// FBSDE_56418690400441
// MI455X (gfx1250) — matrix operand order repaired and checked by reading, not compiled, not run on hardware
//
#include <hip/hip_runtime.h>
#include <hip/hip_bf16.h>
#include <math.h>

typedef __attribute__((ext_vector_type(16))) _Float16 v16h;
typedef __attribute__((ext_vector_type(8)))  float    v8f;
typedef __attribute__((ext_vector_type(4)))  float    f32x4;

union F4x2 { f32x4 f4[2]; v16h h; };

#define BB   2048
#define NN   256
#define DD   2
#define HIDN 256
#define ROWS (BB * NN)          // 524288
#define MU_C    0.1f
#define SIGMA_C 0.2f

// ---------------------------------------------------------------------------
// Kernel 0: convert + transpose W1 weights (256x256 f32, (k,n)) -> f16 (n,k)
// ---------------------------------------------------------------------------
__global__ __launch_bounds__(256) void wprep_kernel(
    const float* __restrict__ fW1, const float* __restrict__ gW1,
    _Float16* __restrict__ fW1t, _Float16* __restrict__ gW1t)
{
    int i = blockIdx.x * 256 + threadIdx.x;          // 0 .. 131071
    const float* W  = (i < 65536) ? fW1  : gW1;
    _Float16*    Wt = (i < 65536) ? fW1t : gW1t;
    int e = i & 65535;
    int n = e >> 8;
    int k = e & 255;
    Wt[e] = (_Float16)W[k * 256 + n];                // Wt[n*256+k] = W[k*256+n]
}

// ---------------------------------------------------------------------------
// Kernel 1: SDE path scan. One thread per batch element b.
// Writes x (B,N,2) to ws and final_value to d_out tail.
// ---------------------------------------------------------------------------
__global__ __launch_bounds__(256) void path_kernel(
    const float* __restrict__ ts, const float* __restrict__ x0,
    const float* __restrict__ eps,
    float* __restrict__ xs, float* __restrict__ fv)
{
    int b = blockIdx.x * 256 + threadIdx.x;
    if (b >= BB) return;
    float xc0 = x0[b * 2 + 0];
    float xc1 = x0[b * 2 + 1];
    size_t base = (size_t)b * NN * 2;
    xs[base + 0] = xc0;
    xs[base + 1] = xc1;
    for (int t = 0; t < NN - 1; ++t) {
        float dt = ts[t + 1] - ts[t];
        float sq = sqrtf(dt);
        float dw0 = eps[base + t * 2 + 0] * sq;
        float dw1 = eps[base + t * 2 + 1] * sq;
        xc0 = xc0 + MU_C * xc0 * dt + SIGMA_C * xc0 * dw0;
        xc1 = xc1 + MU_C * xc1 * dt + SIGMA_C * xc1 * dw1;
        xs[base + (t + 1) * 2 + 0] = xc0;
        xs[base + (t + 1) * 2 + 1] = xc1;
    }
    fv[b] = xc0 * xc0 + xc1 * xc1;
}

// ---------------------------------------------------------------------------
// Kernel 2: fused dual-MLP. 128 threads = 4 waves, 16 rows per wave.
// Layer0: VALU -> f16 h0 in LDS. Layer1: WMMA f32_16x16x32_f16.
// Layer2: folded into accumulator epilogue + lane shuffles.
// ---------------------------------------------------------------------------
__global__ __launch_bounds__(128) void mlp_kernel(
    const float* __restrict__ ts, const float* __restrict__ xs,
    const float* __restrict__ eps,
    const float* __restrict__ fW0, const float* __restrict__ fb0,
    const _Float16* __restrict__ fW1t, const float* __restrict__ fb1,
    const float* __restrict__ fW2, const float* __restrict__ fb2,
    const float* __restrict__ gW0, const float* __restrict__ gb0,
    const _Float16* __restrict__ gW1t, const float* __restrict__ gb1,
    const float* __restrict__ gW2, const float* __restrict__ gb2,
    float* __restrict__ Yout, float* __restrict__ pred)
{
    __shared__ _Float16 h0[64][264];   // 64 rows, stride 264 halves (bank-safe)
    __shared__ float    txs[64][4];

    const int lane = threadIdx.x & 31;
    const int wave = threadIdx.x >> 5;
    const int wrow = wave * 16;                      // local row base of wave
    const int rowW = blockIdx.x * 64 + wrow;         // global row base of wave
    const int lr   = lane & 15;                      // 0..15
    const int hi   = lane >> 4;                      // 0 or 1

    // stage (t, x0, x1) for this wave's 16 rows
    if (lane < 16) {
        int gr = rowW + lane;
        int t  = gr & (NN - 1);
        txs[wrow + lane][0] = ts[t];
        txs[wrow + lane][1] = xs[(size_t)gr * 2 + 0];
        txs[wrow + lane][2] = xs[(size_t)gr * 2 + 1];
    }

    float py[8], pz0[8], pz1[8];
#pragma unroll
    for (int r = 0; r < 8; ++r) { py[r] = 0.f; pz0[r] = 0.f; pz1[r] = 0.f; }

    for (int m = 0; m < 2; ++m) {
        const float*    W0  = m ? gW0  : fW0;
        const float*    b0  = m ? gb0  : fb0;
        const _Float16* W1t = m ? gW1t : fW1t;
        const float*    b1  = m ? gb1  : fb1;
        const float*    W2  = m ? gW2  : fW2;

        __syncthreads();   // txs ready / previous pass done with h0

        // ---- layer 0: h0 = relu(tx @ W0 + b0), f16 into LDS ----
#pragma unroll
        for (int q = 0; q < 4; ++q) {
            int c = 2 * lane + 64 * q;               // even columns 0..254
            float wa0 = W0[c],       wa1 = W0[256 + c],     wa2 = W0[512 + c],     ba = b0[c];
            float wb0 = W0[c + 1],   wb1 = W0[256 + c + 1], wb2 = W0[512 + c + 1], bb = b0[c + 1];
            for (int r16 = 0; r16 < 16; ++r16) {
                float tv  = txs[wrow + r16][0];
                float xv0 = txs[wrow + r16][1];
                float xv1 = txs[wrow + r16][2];
                float ha = fmaf(tv, wa0, fmaf(xv0, wa1, fmaf(xv1, wa2, ba)));
                float hb = fmaf(tv, wb0, fmaf(xv0, wb1, fmaf(xv1, wb2, bb)));
                ha = ha > 0.f ? ha : 0.f;
                hb = hb > 0.f ? hb : 0.f;
                h0[wrow + r16][c]     = (_Float16)ha;
                h0[wrow + r16][c + 1] = (_Float16)hb;
            }
        }
        __syncthreads();

        // ---- A tiles: 16x32 f16 per k-tile, ISA lane layout ----
        v16h A[8];
#pragma unroll
        for (int kt = 0; kt < 8; ++kt) {
            F4x2 u;
            u.f4[0] = *reinterpret_cast<const f32x4*>(&h0[wrow + lr][kt * 32 + hi * 8]);
            u.f4[1] = *reinterpret_cast<const f32x4*>(&h0[wrow + lr][kt * 32 + 16 + hi * 8]);
            A[kt] = u.h;
        }

        // ---- layer 1 (WMMA) + layer 2 (fused epilogue) ----
        for (int nt = 0; nt < 16; ++nt) {
            int col = nt * 16 + lr;
            const _Float16* wp = W1t + (size_t)col * 256 + hi * 8;
            v8f acc = {0.f, 0.f, 0.f, 0.f, 0.f, 0.f, 0.f, 0.f};
#pragma unroll
            for (int kt = 0; kt < 8; ++kt) {
                F4x2 bt;
                bt.f4[0] = *reinterpret_cast<const f32x4*>(wp + kt * 32);
                bt.f4[1] = *reinterpret_cast<const f32x4*>(wp + kt * 32 + 16);
                acc = __builtin_amdgcn_wmma_f32_16x16x32_f16(
                    false, A[kt], false, bt.h, (short)0, acc, false, false);
            }
            float bias = b1[col];
            if (m == 0) {
                float w2 = W2[col];
#pragma unroll
                for (int r = 0; r < 8; ++r) {
                    float h = acc[r] + bias;
                    h = h > 0.f ? h : 0.f;
                    py[r] = fmaf(h, w2, py[r]);
                }
            } else {
                float wz0 = W2[col * 2 + 0];
                float wz1 = W2[col * 2 + 1];
#pragma unroll
                for (int r = 0; r < 8; ++r) {
                    float h = acc[r] + bias;
                    h = h > 0.f ? h : 0.f;
                    pz0[r] = fmaf(h, wz0, pz0[r]);
                    pz1[r] = fmaf(h, wz1, pz1[r]);
                }
            }
        }
    }

    // ---- reduce partial dots across the 16 lanes of each half-wave ----
#pragma unroll
    for (int r = 0; r < 8; ++r) {
#pragma unroll
        for (int msk = 1; msk < 16; msk <<= 1) {
            py[r]  += __shfl_xor(py[r],  msk, 32);
            pz0[r] += __shfl_xor(pz0[r], msk, 32);
            pz1[r] += __shfl_xor(pz1[r], msk, 32);
        }
    }

    if (lr == 0) {
        float yb  = fb2[0];
        float zb0 = gb2[0], zb1 = gb2[1];
#pragma unroll
        for (int r = 0; r < 8; ++r) {
            int grow = rowW + hi * 8 + r;            // global row (b*N + t)
            int t = grow & (NN - 1);
            float Y = py[r] + yb;
            Yout[grow] = Y;
            float dw0 = 0.f, dw1 = 0.f;
            if (t < NN - 1) {
                float sq = sqrtf(ts[t + 1] - ts[t]);
                dw0 = eps[(size_t)grow * 2 + 0] * sq;
                dw1 = eps[(size_t)grow * 2 + 1] * sq;
            }
            float z0 = pz0[r] + zb0;
            float z1 = pz1[r] + zb1;
            pred[grow] = fmaf(z0, dw0, fmaf(z1, dw1, Y));
        }
    }
}

// ---------------------------------------------------------------------------
// Kernel 3: per-block partial sums of squared errors (deterministic).
// ---------------------------------------------------------------------------
__global__ __launch_bounds__(512) void loss_part_kernel(
    const float* __restrict__ Yout, const float* __restrict__ fv,
    const float* __restrict__ pred, float* __restrict__ partials)
{
    __shared__ float sm[512];
    int r = blockIdx.x * 512 + threadIdx.x;          // exactly covers ROWS
    int t = r & (NN - 1);
    float target = (t < NN - 1) ? Yout[r + 1] : fv[r >> 8];
    float d = pred[r] - target;
    sm[threadIdx.x] = d * d;
    __syncthreads();
    for (int s = 256; s > 0; s >>= 1) {
        if (threadIdx.x < s) sm[threadIdx.x] += sm[threadIdx.x + s];
        __syncthreads();
    }
    if (threadIdx.x == 0) partials[blockIdx.x] = sm[0];
}

__global__ __launch_bounds__(256) void loss_final_kernel(
    const float* __restrict__ partials, float* __restrict__ loss)
{
    __shared__ float sm[256];
    float s = 0.f;
    for (int i = threadIdx.x; i < 1024; i += 256) s += partials[i];
    sm[threadIdx.x] = s;
    __syncthreads();
    for (int st = 128; st > 0; st >>= 1) {
        if (threadIdx.x < st) sm[threadIdx.x] += sm[threadIdx.x + st];
        __syncthreads();
    }
    if (threadIdx.x == 0) loss[0] = sm[0] * (1.0f / (float)BB);
}

// ---------------------------------------------------------------------------
extern "C" void kernel_launch(void* const* d_in, const int* in_sizes, int n_in,
                              void* d_out, int out_size, void* d_ws, size_t ws_size,
                              hipStream_t stream) {
    const float* ts  = (const float*)d_in[0];
    const float* x0  = (const float*)d_in[1];
    const float* eps = (const float*)d_in[2];
    const float* fW0 = (const float*)d_in[3];
    const float* fb0 = (const float*)d_in[4];
    const float* fW1 = (const float*)d_in[5];
    const float* fb1 = (const float*)d_in[6];
    const float* fW2 = (const float*)d_in[7];
    const float* fb2 = (const float*)d_in[8];
    const float* gW0 = (const float*)d_in[9];
    const float* gb0 = (const float*)d_in[10];
    const float* gW1 = (const float*)d_in[11];
    const float* gb1 = (const float*)d_in[12];
    const float* gW2 = (const float*)d_in[13];
    const float* gb2 = (const float*)d_in[14];

    float* out  = (float*)d_out;
    float* loss = out;                       // [0]
    float* Yout = out + 1;                   // [1 .. 1+ROWS)
    float* fv   = out + 1 + ROWS;            // [.. +B)

    // workspace layout (16B-aligned pieces)
    char* ws = (char*)d_ws;
    size_t offX    = 0;                                  // x: B*N*2 f32 = 4 MB
    size_t offPred = offX + (size_t)ROWS * 2 * 4;        // pred: ROWS f32 = 2 MB
    size_t offFW1t = offPred + (size_t)ROWS * 4;         // f16 65536 = 128 KB
    size_t offGW1t = offFW1t + 65536 * 2;
    size_t offPart = offGW1t + 65536 * 2;                // 1024 f32

    float*    xs    = (float*)(ws + offX);
    float*    pred  = (float*)(ws + offPred);
    _Float16* fW1t  = (_Float16*)(ws + offFW1t);
    _Float16* gW1t  = (_Float16*)(ws + offGW1t);
    float*    parts = (float*)(ws + offPart);

    wprep_kernel<<<512, 256, 0, stream>>>(fW1, gW1, fW1t, gW1t);
    path_kernel<<<(BB + 255) / 256, 256, 0, stream>>>(ts, x0, eps, xs, fv);
    mlp_kernel<<<ROWS / 64, 128, 0, stream>>>(
        ts, xs, eps,
        fW0, fb0, fW1t, fb1, fW2, fb2,
        gW0, gb0, gW1t, gb1, gW2, gb2,
        Yout, pred);
    loss_part_kernel<<<ROWS / 512, 512, 0, stream>>>(Yout, fv, pred, parts);
    loss_final_kernel<<<1, 256, 0, stream>>>(parts, loss);
}
